// MANN_75273596829746
// MI455X (gfx1250) — matrix operand order repaired and checked by reading, not compiled, not run on hardware
//
#include <hip/hip_runtime.h>
#include <hip/hip_bf16.h>

typedef __attribute__((ext_vector_type(16))) _Float16 v16h;
typedef __attribute__((ext_vector_type(8)))  _Float16 v8h;
typedef __attribute__((ext_vector_type(8)))  float    v8f;

#define B_SZ 512
#define T_SZ 512
#define I_SZ 20
#define H_SZ 128

// ---------------- workspace layout (bytes) ----------------
// xT      : [T][B][32] f16 (time-major, K padded 20->32)   16,777,216
// whh_sw  : WMMA B-fragments of W_hh^T (4 kt x 32 nt)         131,072
// wih_sw  : WMMA B-fragments of W_ih^T (1 kt x 32 nt)          32,768
// w1_sw   : WMMA B-fragments of W1^T   (4 kt x 4 nt)           16,384
static const size_t XT_OFF  = 0;
static const size_t WHH_OFF = (size_t)T_SZ * B_SZ * 32 * 2;     // 16777216
static const size_t WIH_OFF = WHH_OFF + 4 * 32 * 512 * 2;       // +131072
static const size_t W1_OFF  = WIH_OFF + 32 * 512 * 2;           // +32768

// ============ prep: x [B,T,20] f32 -> xT [T,B,32] f16 (zero-padded) ============
__global__ void prep_x(const float* __restrict__ x, _Float16* __restrict__ xT) {
  int i = blockIdx.x * 256 + threadIdx.x;        // 512*512*32 = 8388608 exact
  int k = i & 31;
  int b = (i >> 5) & 511;
  int t = i >> 14;
  float v = (k < I_SZ) ? x[((size_t)b * T_SZ + t) * I_SZ + k] : 0.f;
  xT[i] = (_Float16)v;
}

// ============ prep: swizzle weights into WMMA B-fragment order ============
// B-fragment (16x16x32 f16): lane L holds N=n0+(L&15), element e holds
// K = k0 + (e&7) + 8*(L>>4) + 16*(e>>3).  Flat: frag[((tile)*32 + L)*16 + e]
__global__ void prep_w(const float* __restrict__ Wih, const float* __restrict__ Whh,
                       const float* __restrict__ W1,
                       _Float16* __restrict__ whh_sw, _Float16* __restrict__ wih_sw,
                       _Float16* __restrict__ w1_sw) {
  int i = blockIdx.x * 256 + threadIdx.x;
  if (i < 65536) {                                  // W_hh: 4 kt x 32 nt
    int e = i & 15, L = (i >> 4) & 31, tl = i >> 9;
    int nt = tl & 31, kt = tl >> 5;
    whh_sw[i] = (_Float16)Whh[(nt * 16 + (L & 15)) * H_SZ + kt * 32 + (e & 7) + 8 * (L >> 4) + 16 * (e >> 3)];
  } else if (i < 65536 + 16384) {                   // W_ih: 32 nt, K padded to 32
    int j = i - 65536;
    int e = j & 15, L = (j >> 4) & 31, nt = j >> 9;
    int kk = (e & 7) + 8 * (L >> 4) + 16 * (e >> 3);
    float v = (kk < I_SZ) ? Wih[(nt * 16 + (L & 15)) * I_SZ + kk] : 0.f;
    wih_sw[j] = (_Float16)v;
  } else if (i < 65536 + 16384 + 8192) {            // W1: 4 kt x 4 nt
    int j = i - 65536 - 16384;
    int e = j & 15, L = (j >> 4) & 31, tl = j >> 9;
    int nt = tl & 3, kt = tl >> 2;
    w1_sw[j] = (_Float16)W1[(nt * 16 + (L & 15)) * H_SZ + kt * 32 + (e & 7) + 8 * (L >> 4) + 16 * (e >> 3)];
  }
}

__device__ __forceinline__ float sigf(float x) {
  return __builtin_amdgcn_rcpf(1.0f + __expf(-x));
}
__device__ __forceinline__ float tanhf_fast(float x) {
  return __builtin_fmaf(2.0f, sigf(2.0f * x), -1.0f);
}

// A-fragment gather (16x16x32 f16 WMMA A layout): lane (m,hi) takes
// K = {8hi..8hi+7} and {8hi+16..8hi+23} of row m -> two 16B loads.
__device__ __forceinline__ v16h load_frag_a(const _Float16* p) {
  v8h a = *(const v8h*)p;
  v8h b = *(const v8h*)(p + 16);
  return __builtin_shufflevector(a, b, 0,1,2,3,4,5,6,7,8,9,10,11,12,13,14,15);
}

#define H_STRIDE 136   // f16, padded: 272B rows -> conflict-free fragment loads
#define HD_STRIDE 68   // f32, padded

// ============ persistent scan kernel: 32 blocks x 16 batch rows ============
// Wave w owns gate N-tiles {w, w+8, w+16, w+24}: accumulator tile i holds
// gate i (i/f/g/o) for hidden columns 16w..16w+15 -> cell math stays in
// registers; cell state c lives in 8 VGPRs per lane. Only h goes via LDS.
// Head GEMM is folded into phase A one step delayed (reuses ah fragments),
// so each timestep needs only 2 barriers.
__global__ void __launch_bounds__(256, 1)
lstm_scan(const _Float16* __restrict__ xT,
          const _Float16* __restrict__ whh_g,
          const _Float16* __restrict__ wih_g,
          const _Float16* __restrict__ w1_g,
          const float* __restrict__ b_ih, const float* __restrict__ b_hh,
          const float* __restrict__ b1,   const float* __restrict__ W2,
          const float* __restrict__ b2,   float* __restrict__ out) {
  __shared__ _Float16 sH[16 * H_STRIDE];      // hidden state, f16
  __shared__ float    sHdn[16 * HD_STRIDE];   // head hidden, f32
  __shared__ float    sW2[256];
  __shared__ float    sB2[4];

  const int tid = threadIdx.x;
  const int w   = tid >> 5;     // wave 0..7
  const int L   = tid & 31;
  const int m   = L & 15;
  const int hi  = L >> 4;
  const int b0  = blockIdx.x * 16;

  // ---- init: small LDS staging, zero h ----
  if (tid < 256) sW2[tid] = W2[tid];
  if (tid < 4)   sB2[tid] = b2[tid];
  for (int i = tid; i < 16 * H_STRIDE; i += 256) sH[i] = (_Float16)0.f;

  // ---- register-resident weight fragments ----
  v16h wihf[4];          // W_ih B-frags, N-tiles w+8i
  v16h whhf[4][4];       // W_hh B-frags [kt][i]
  #pragma unroll
  for (int i = 0; i < 4; ++i) {
    int nt = w + 8 * i;
    wihf[i] = *(const v16h*)(wih_g + ((size_t)(nt * 32 + L)) * 16);
    #pragma unroll
    for (int kt = 0; kt < 4; ++kt)
      whhf[kt][i] = *(const v16h*)(whh_g + ((size_t)((kt * 32 + nt) * 32 + L)) * 16);
  }
  v16h w1f[4];           // W1 B-frags for waves 0..3
  if (w < 4) {
    #pragma unroll
    for (int kt = 0; kt < 4; ++kt)
      w1f[kt] = *(const v16h*)(w1_g + ((size_t)((kt * 4 + w) * 32 + L)) * 16);
  }

  // lane-fixed constants: this lane owns hidden column j for the whole run
  const int j = (w << 4) | m;
  const float bi = b_ih[j]           + b_hh[j];
  const float bf = b_ih[H_SZ + j]    + b_hh[H_SZ + j];
  const float bg = b_ih[2*H_SZ + j]  + b_hh[2*H_SZ + j];
  const float bo = b_ih[3*H_SZ + j]  + b_hh[3*H_SZ + j];
  const float b1v = (w < 4) ? b1[w * 16 + m] : 0.f;

  float cst[8] = {0.f, 0.f, 0.f, 0.f, 0.f, 0.f, 0.f, 0.f};  // cell state

  __syncthreads();

  // software-pipelined x fragment (global)
  v16h ax = load_frag_a(xT + (((size_t)b0 + m) * 32 + 8 * hi));

  for (int t = 0; t < T_SZ; ++t) {
    // ---- phase A: gates(t) = x_t*Wih^T + h(t-1)*Whh^T, and (waves 0-3)
    //      head hdn(t-1) = relu(h(t-1)*W1^T + b1), reusing the ah frags ----
    v16h ah[4];
    #pragma unroll
    for (int kt = 0; kt < 4; ++kt)
      ah[kt] = load_frag_a(sH + m * H_STRIDE + kt * 32 + 8 * hi);

    v8f acc[4] = {};
    #pragma unroll
    for (int i = 0; i < 4; ++i)
      acc[i] = __builtin_amdgcn_wmma_f32_16x16x32_f16(
          false, ax, false, wihf[i], (short)0, acc[i], false, false);
    #pragma unroll
    for (int kt = 0; kt < 4; ++kt)
      #pragma unroll
      for (int i = 0; i < 4; ++i)
        acc[i] = __builtin_amdgcn_wmma_f32_16x16x32_f16(
            false, ah[kt], false, whhf[kt][i], (short)0, acc[i], false, false);

    if (w < 4) {            // delayed head on h(t-1); at t=0 h(-1)=0 (unused)
      v8f ha = {};
      #pragma unroll
      for (int kt = 0; kt < 4; ++kt)
        ha = __builtin_amdgcn_wmma_f32_16x16x32_f16(
            false, ah[kt], false, w1f[kt], (short)0, ha, false, false);
      int col = w * 16 + m;
      #pragma unroll
      for (int r = 0; r < 8; ++r) {
        float v = ha[r] + b1v;
        sHdn[(r + 8 * hi) * HD_STRIDE + col] = v > 0.f ? v : 0.f;
      }
    }

    // prefetch next timestep's x fragment (hides global latency under B)
    {
      int t2 = (t + 1 < T_SZ) ? t + 1 : t;
      ax = load_frag_a(xT + (((size_t)t2 * B_SZ + b0 + m) * 32 + 8 * hi));
    }
    __syncthreads();   // sH reads + sHdn writes complete

    // ---- phase B: LSTM cell (registers) -> h(t); waves 4-5: preds(t-1) ----
    #pragma unroll
    for (int r = 0; r < 8; ++r) {
      float gi = acc[0][r] + bi;
      float gf = acc[1][r] + bf;
      float gg = acc[2][r] + bg;
      float go = acc[3][r] + bo;
      float cn = sigf(gf) * cst[r] + sigf(gi) * tanhf_fast(gg);
      float hn = sigf(go) * tanhf_fast(cn);
      cst[r] = cn;
      sH[(r + 8 * hi) * H_STRIDE + j] = (_Float16)hn;
    }
    if (t && (w == 4 || w == 5)) {
      int o = (w - 4) * 2 + hi;
      float sum = sB2[o];
      const float4* hr = (const float4*)(sHdn + m * HD_STRIDE);
      const float4* wr = (const float4*)(sW2 + o * 64);
      #pragma unroll
      for (int kk = 0; kk < 16; ++kk) {
        float4 a = hr[kk], bb = wr[kk];
        sum += a.x * bb.x + a.y * bb.y + a.z * bb.z + a.w * bb.w;
      }
      out[((size_t)(b0 + m) * T_SZ + (t - 1)) * 4 + o] = sum;
    }
    __syncthreads();   // h(t) visible; sHdn reads complete
  }

  // ---- epilogue: head + preds for h(T-1) -> out[T-1] ----
  if (w < 4) {
    v8f ha = {};
    #pragma unroll
    for (int kt = 0; kt < 4; ++kt) {
      v16h ahn = load_frag_a(sH + m * H_STRIDE + kt * 32 + 8 * hi);
      ha = __builtin_amdgcn_wmma_f32_16x16x32_f16(
          false, ahn, false, w1f[kt], (short)0, ha, false, false);
    }
    int col = w * 16 + m;
    #pragma unroll
    for (int r = 0; r < 8; ++r) {
      float v = ha[r] + b1v;
      sHdn[(r + 8 * hi) * HD_STRIDE + col] = v > 0.f ? v : 0.f;
    }
  }
  __syncthreads();
  if (w < 2) {
    int o = w * 2 + hi;
    float sum = sB2[o];
    const float4* hr = (const float4*)(sHdn + m * HD_STRIDE);
    const float4* wr = (const float4*)(sW2 + o * 64);
    #pragma unroll
    for (int kk = 0; kk < 16; ++kk) {
      float4 a = hr[kk], bb = wr[kk];
      sum += a.x * bb.x + a.y * bb.y + a.z * bb.z + a.w * bb.w;
    }
    out[((size_t)(b0 + m) * T_SZ + (T_SZ - 1)) * 4 + o] = sum;
  }
}

extern "C" void kernel_launch(void* const* d_in, const int* in_sizes, int n_in,
                              void* d_out, int out_size, void* d_ws, size_t ws_size,
                              hipStream_t stream) {
  (void)in_sizes; (void)n_in; (void)out_size; (void)ws_size;
  const float* x    = (const float*)d_in[0];
  // d_in[1] contexts, d_in[2] lengths: unused by the reference forward
  const float* W_ih = (const float*)d_in[3];
  const float* W_hh = (const float*)d_in[4];
  const float* b_ih = (const float*)d_in[5];
  const float* b_hh = (const float*)d_in[6];
  const float* W1   = (const float*)d_in[7];
  const float* b1   = (const float*)d_in[8];
  const float* W2   = (const float*)d_in[9];
  const float* b2   = (const float*)d_in[10];
  float* out = (float*)d_out;

  char* ws = (char*)d_ws;
  _Float16* xT     = (_Float16*)(ws + XT_OFF);
  _Float16* whh_sw = (_Float16*)(ws + WHH_OFF);
  _Float16* wih_sw = (_Float16*)(ws + WIH_OFF);
  _Float16* w1_sw  = (_Float16*)(ws + W1_OFF);

  prep_x<<<32768, 256, 0, stream>>>(x, xT);
  prep_w<<<352, 256, 0, stream>>>(W_ih, W_hh, W1, whh_sw, wih_sw, w1_sw);
  lstm_scan<<<32, 256, 0, stream>>>(xT, whh_sw, wih_sw, w1_sw,
                                    b_ih, b_hh, b1, W2, b2, out);
}
